// AMMVGG_62904091018027
// MI455X (gfx1250) — matrix operand order repaired and checked by reading, not compiled, not run on hardware
//
#include <hip/hip_runtime.h>
#include <hip/hip_bf16.h>
#include <stdint.h>

// ---------------------------------------------------------------------------
// CDNA5 (gfx1250) AMM-VGG11 inference.
// Dominant op: per-AMM-layer GEMM  attn[N*L, C*16] x lut[C*16, outC]
// mapped onto v_wmma_f32_16x16x32_f16 (f16 in, f32 accumulate).
// ---------------------------------------------------------------------------

typedef _Float16 half_t;
typedef __attribute__((ext_vector_type(16))) _Float16 v16h;
typedef __attribute__((ext_vector_type(8)))  _Float16 v8h;
typedef __attribute__((ext_vector_type(8)))  float    v8f;

union HF16 {            // 16 halves viewed as two 16B chunks (for fragment loads)
    v16h v;
    v8h  h2[2];
};
union PK16 {            // 16 halves viewed for packed 16B stores
    v8h     h8[2];
    half_t  h[16];
};
union CV32 {            // one dword as two halves
    uint32_t u;
    half_t   h[2];
};

#define EPS 1e-5f

// ---------------------------------------------------------------------------
// Kernel 1: first conv 3->64 on 32x32, SAME pad, + bias + BN + ReLU (NCHW f32)
// ---------------------------------------------------------------------------
__global__ __launch_bounds__(256)
void conv1_bn_relu_kernel(const float* __restrict__ x, const float* __restrict__ w,
                          const float* __restrict__ bconv,
                          const float* __restrict__ g, const float* __restrict__ be,
                          const float* __restrict__ mu, const float* __restrict__ va,
                          float* __restrict__ out)
{
    const int total = 128 * 64 * 32 * 32;
    int tid = blockIdx.x * blockDim.x + threadIdx.x;
    if (tid >= total) return;
    int p    = tid & 1023;          // pixel
    int rest = tid >> 10;
    int oc   = rest & 63;
    int b    = rest >> 6;
    int y  = p >> 5;
    int xw = p & 31;

    float sum = bconv[oc];
    #pragma unroll
    for (int c = 0; c < 3; ++c) {
        const float* xc = x + ((size_t)(b * 3 + c) << 10);
        const float* wc = w + ((oc * 3 + c) * 9);
        #pragma unroll
        for (int ky = 0; ky < 3; ++ky) {
            int yy = y + ky - 1;
            if (yy < 0 || yy > 31) continue;
            #pragma unroll
            for (int kx = 0; kx < 3; ++kx) {
                int xx = xw + kx - 1;
                if (xx < 0 || xx > 31) continue;
                sum += wc[ky * 3 + kx] * xc[(yy << 5) + xx];
            }
        }
    }
    float sc = g[oc] * rsqrtf(va[oc] + EPS);
    float v  = sum * sc + (be[oc] - mu[oc] * sc);
    v = v > 0.f ? v : 0.f;
    out[(((size_t)(b * 64 + oc)) << 10) + p] = v;
}

// ---------------------------------------------------------------------------
// Kernel 2: 2x2 max pool (NCHW f32)
// ---------------------------------------------------------------------------
__global__ __launch_bounds__(256)
void maxpool_kernel(const float* __restrict__ in, float* __restrict__ out,
                    int C, int Hout, int total)
{
    int tid = blockIdx.x * blockDim.x + threadIdx.x;
    if (tid >= total) return;
    int Wout = Hout;
    int l  = tid % (Hout * Wout);
    int c  = (tid / (Hout * Wout)) % C;
    int b  = tid / (Hout * Wout * C);
    int y  = l / Wout;
    int xw = l % Wout;
    int Hin = Hout * 2;
    const float* p = in + (size_t)(b * C + c) * Hin * Hin;
    int y2 = y * 2, x2 = xw * 2;
    float m0 = fmaxf(p[y2 * Hin + x2],       p[y2 * Hin + x2 + 1]);
    float m1 = fmaxf(p[(y2 + 1) * Hin + x2], p[(y2 + 1) * Hin + x2 + 1]);
    out[tid] = fmaxf(m0, m1);
}

// ---------------------------------------------------------------------------
// Kernel 3: f32 -> f16 LUT conversion
// ---------------------------------------------------------------------------
__global__ __launch_bounds__(256)
void cvt_f16_kernel(const float* __restrict__ in, half_t* __restrict__ out, int n)
{
    int tid = blockIdx.x * blockDim.x + threadIdx.x;
    if (tid < n) out[tid] = (half_t)in[tid];
}

// ---------------------------------------------------------------------------
// Kernel 4: AMM distance + softmax -> attn row-major f16 matrix [N*L, C*16].
// One thread per (batch, codebook==channel, pixel). Packs 16 weights and
// stores them as two 16-byte vectors (row start is 32B aligned).
// ---------------------------------------------------------------------------
__global__ __launch_bounds__(256)
void amm_attn_kernel(const float* __restrict__ x, const float* __restrict__ cent,
                     half_t* __restrict__ attn,
                     int C, int Cshift, int H, int Wshift, int Lshift, int Kd,
                     int total)
{
    int tid = blockIdx.x * blockDim.x + threadIdx.x;
    if (tid >= total) return;
    const int L = 1 << Lshift;
    int l    = tid & (L - 1);
    int rest = tid >> Lshift;
    int n    = rest & (C - 1);
    int b    = rest >> Cshift;
    int y  = l >> Wshift;
    int xw = l & (H - 1);

    // 3x3 patch of channel n (zero padded): feature order (ky, kx)
    float a[9];
    const float* xc = x + ((size_t)(b * C + n) << Lshift);
    #pragma unroll
    for (int ky = 0; ky < 3; ++ky) {
        #pragma unroll
        for (int kx = 0; kx < 3; ++kx) {
            int yy = y + ky - 1, xx = xw + kx - 1;
            bool ok = (yy >= 0) && (yy < H) && (xx >= 0) && (xx < H);
            a[ky * 3 + kx] = ok ? xc[(yy << Wshift) + xx] : 0.f;
        }
    }
    float a2 = 0.f;
    #pragma unroll
    for (int j = 0; j < 9; ++j) a2 += a[j] * a[j];

    const float* cp = cent + (size_t)n * 16 * 9;
    float s[16];
    float m = -3.0e38f;
    #pragma unroll
    for (int k = 0; k < 16; ++k) {
        float cross = 0.f, c2 = 0.f;
        #pragma unroll
        for (int j = 0; j < 9; ++j) {
            float cv = cp[k * 9 + j];
            cross += a[j] * cv;
            c2    += cv * cv;
        }
        s[k] = -(a2 - 2.f * cross + c2);   // TEMP == 1
        m = fmaxf(m, s[k]);
    }
    float sum = 0.f, e[16];
    #pragma unroll
    for (int k = 0; k < 16; ++k) { e[k] = __expf(s[k] - m); sum += e[k]; }
    float inv = 1.f / sum;

    PK16 pk;
    #pragma unroll
    for (int k = 0; k < 16; ++k) pk.h[k] = (half_t)(e[k] * inv);

    half_t* dst = attn + (size_t)(b * L + l) * Kd + n * 16;
    *((v8h*)dst)     = pk.h8[0];
    *((v8h*)dst + 1) = pk.h8[1];
}

// ---------------------------------------------------------------------------
// Kernel 5: WMMA GEMM  D[M, outC] = A[M, Kd] x B[Kd, outC]  (f16 -> f32 acc)
// fused bias + BN + ReLU, scattered to NCHW.
// Block: 256 threads = 8 waves in a 2(M) x 4(N) grid -> 32x64 tile.
// A/B staged through LDS; fragments obey the CDNA5 16-bit WMMA VGPR layouts:
//   A 16x32: lane&15 = row, lane>>4 picks K-octets {0..7,16..23}/{8..15,24..31}
//   B 32x16: lane&15 = col, lane>>4 picks K-octets {0..7,16..23}/{8..15,24..31}
//   C/D 16x16 f32: vgpr r -> row (lane>>4)*8 + r, col = lane&15
// ---------------------------------------------------------------------------
__global__ __launch_bounds__(256, 2)
void amm_gemm_wmma_kernel(const half_t* __restrict__ A, const half_t* __restrict__ B,
                          const float* __restrict__ bias,
                          const float* __restrict__ g, const float* __restrict__ be,
                          const float* __restrict__ mu, const float* __restrict__ va,
                          float* __restrict__ out,
                          int Kd, int outC, int Lshift)
{
    __shared__ __align__(16) half_t As[32 * 40];   // 32 rows x 32 K (stride 40)
    __shared__ __align__(16) half_t Bt[64 * 40];   // 64 cols x 32 K (stride 40, transposed)

    const int tid   = threadIdx.x;
    const int lane  = tid & 31;
    const int wave  = tid >> 5;
    const int waveM = wave >> 2;       // 0..1
    const int waveN = wave & 3;        // 0..3
    const int rowBase = blockIdx.y * 32;
    const int colBase = blockIdx.x * 64;
    const int Lmask = (1 << Lshift) - 1;

    v8f acc = {};

    for (int k0 = 0; k0 < Kd; k0 += 32) {
        // ---- cooperative A tile load: 32 rows x 32 halves (512 dwords) ----
        #pragma unroll
        for (int i = 0; i < 2; ++i) {
            int e = tid + i * 256;
            int r = e >> 4;                 // 16 dwords per row
            int u = e & 15;
            const uint32_t* src =
                (const uint32_t*)(A + (size_t)(rowBase + r) * Kd + k0 + u * 2);
            *(uint32_t*)(As + r * 40 + u * 2) = *src;
        }
        // ---- cooperative B tile load, transposed into LDS ----
        #pragma unroll
        for (int i = 0; i < 4; ++i) {
            int e = tid + i * 256;
            int k = e >> 5;                 // 32 dwords per K row
            int u = e & 31;                 // column pair
            CV32 cv;
            cv.u = *(const uint32_t*)(B + (size_t)(k0 + k) * outC + colBase + u * 2);
            Bt[(u * 2 + 0) * 40 + k] = cv.h[0];
            Bt[(u * 2 + 1) * 40 + k] = cv.h[1];
        }
        // prefetch next K slice while we compute (global_prefetch_b8)
        if (k0 + 32 < Kd) {
            __builtin_prefetch(A + (size_t)(rowBase + (tid >> 3)) * Kd + k0 + 32, 0, 1);
            __builtin_prefetch(B + (size_t)(k0 + 32 + (tid >> 3)) * outC + colBase, 0, 1);
        }
        __syncthreads();

        // ---- fragment loads from LDS ----
        HF16 af, bf;
        {
            const half_t* Ar = As + (waveM * 16 + (lane & 15)) * 40;
            const int kA = (lane >> 4) << 3;              // 0 or 8
            af.h2[0] = *(const v8h*)(Ar + kA);            // K = kA .. kA+7
            af.h2[1] = *(const v8h*)(Ar + kA + 16);       // K = kA+16 .. kA+23

            const half_t* Bc = Bt + (waveN * 16 + (lane & 15)) * 40;
            const int kB = (lane >> 4) << 3;              // 0 or 8
            bf.h2[0] = *(const v8h*)(Bc + kB);            // K = kB .. kB+7
            bf.h2[1] = *(const v8h*)(Bc + kB + 16);       // K = kB+16 .. kB+23
        }

        acc = __builtin_amdgcn_wmma_f32_16x16x32_f16(
                  /*neg_a=*/false, af.v, /*neg_b=*/false, bf.v,
                  /*c_mod=*/(short)0, acc, /*reuse_a=*/false, /*reuse_b=*/false);
        __syncthreads();
    }

    // ---- fused bias + BN + ReLU epilogue, scatter to NCHW ----
    const int col = colBase + waveN * 16 + (lane & 15);
    const float sc = g[col] * rsqrtf(va[col] + EPS);
    const float sh = be[col] - mu[col] * sc;
    const float bs = bias[col];
    const int rbase = rowBase + waveM * 16 + ((lane >> 4) << 3);
    #pragma unroll
    for (int r = 0; r < 8; ++r) {
        int row = rbase + r;                 // row = b*L + l
        int b   = row >> Lshift;
        int l   = row & Lmask;
        float v = (acc[r] + bs) * sc + sh;
        v = v > 0.f ? v : 0.f;
        out[(((size_t)b * outC + col) << Lshift) | (unsigned)l] = v;
    }
}

// ---------------------------------------------------------------------------
// Kernel 6: global-avg-pool (H=W=1 -> identity) + FC 512 -> 10
// ---------------------------------------------------------------------------
__global__ __launch_bounds__(256)
void fc_kernel(const float* __restrict__ act, const float* __restrict__ wfc,
               const float* __restrict__ bfc, float* __restrict__ out)
{
    int tid = blockIdx.x * blockDim.x + threadIdx.x;
    if (tid >= 128 * 10) return;
    int o = tid % 10;
    int b = tid / 10;
    float s = bfc[o];
    const float* a = act + (size_t)b * 512;
    const float* w = wfc + (size_t)o * 512;
    #pragma unroll 8
    for (int c = 0; c < 512; ++c) s += a[c] * w[c];
    out[b * 10 + o] = s;
}

// ---------------------------------------------------------------------------
// Host orchestration
// ---------------------------------------------------------------------------
struct AmmDesc { int ci, co, H, pidx, pool_after; };

static int ilog2(int v) { int s = 0; while ((1 << s) < v) ++s; return s; }

extern "C" void kernel_launch(void* const* d_in, const int* in_sizes, int n_in,
                              void* d_out, int out_size, void* d_ws, size_t ws_size,
                              hipStream_t stream)
{
    (void)in_sizes; (void)n_in; (void)out_size; (void)ws_size;

    const float* x = (const float*)d_in[0];

    // workspace layout (bytes)
    const size_t OFF_ACT0 = 0;                       // 128*64*32*32*4 = 32 MiB
    const size_t OFF_ACT1 = 33554432;
    const size_t OFF_ATTN = 67108864;                // up to 32768*1024*2 = 64 MiB
    const size_t OFF_LUT  = 134217728;               // up to 512*16*512*2 = 8 MiB
    float*  actA = (float*)((char*)d_ws + OFF_ACT0);
    float*  actB = (float*)((char*)d_ws + OFF_ACT1);
    half_t* attn = (half_t*)((char*)d_ws + OFF_ATTN);
    half_t* lutf = (half_t*)((char*)d_ws + OFF_LUT);

    // ---- conv1 + BN + ReLU -> actA [128,64,32,32] ----
    {
        int total = 128 * 64 * 32 * 32;
        conv1_bn_relu_kernel<<<(total + 255) / 256, 256, 0, stream>>>(
            x, (const float*)d_in[1], (const float*)d_in[2],
            (const float*)d_in[3], (const float*)d_in[4],
            (const float*)d_in[5], (const float*)d_in[6], actA);
    }
    // ---- pool -> actB [128,64,16,16] ----
    {
        int total = 128 * 64 * 16 * 16;
        maxpool_kernel<<<(total + 255) / 256, 256, 0, stream>>>(actA, actB, 64, 16, total);
    }

    float* cur   = actB;
    float* other = actA;
    int curH = 16;

    static const AmmDesc layers[7] = {
        { 64, 128, 16,  7, 1},
        {128, 256,  8, 14, 0},
        {256, 256,  8, 21, 1},
        {256, 512,  4, 28, 0},
        {512, 512,  4, 35, 1},
        {512, 512,  2, 42, 0},
        {512, 512,  2, 49, 1},
    };

    for (int li = 0; li < 7; ++li) {
        const AmmDesc& d = layers[li];
        const float* cent = (const float*)d_in[d.pidx + 0];
        const float* lut  = (const float*)d_in[d.pidx + 1];
        const float* bias = (const float*)d_in[d.pidx + 2];
        const float* g    = (const float*)d_in[d.pidx + 3];
        const float* be   = (const float*)d_in[d.pidx + 4];
        const float* mu   = (const float*)d_in[d.pidx + 5];
        const float* va   = (const float*)d_in[d.pidx + 6];

        const int H = d.H, C = d.ci, outC = d.co;
        const int L = H * H;
        const int Kd = C * 16;
        const int M = 128 * L;
        const int Lshift = ilog2(L);

        // distances + softmax -> attn f16 [M, Kd]
        {
            int total = 128 * C * L;
            amm_attn_kernel<<<(total + 255) / 256, 256, 0, stream>>>(
                cur, cent, attn, C, ilog2(C), H, ilog2(H), Lshift, Kd, total);
        }
        // LUT -> f16 [Kd, outC]
        {
            int n = Kd * outC;
            cvt_f16_kernel<<<(n + 255) / 256, 256, 0, stream>>>(lut, lutf, n);
        }
        // WMMA GEMM + bias + BN + ReLU -> other (NCHW)
        {
            dim3 grid(outC / 64, M / 32);
            amm_gemm_wmma_kernel<<<grid, 256, 0, stream>>>(
                attn, lutf, bias, g, be, mu, va, other, Kd, outC, Lshift);
        }
        { float* t = cur; cur = other; other = t; }

        if (d.pool_after) {
            int Hout = curH / 2;
            int total = 128 * outC * Hout * Hout;
            maxpool_kernel<<<(total + 255) / 256, 256, 0, stream>>>(cur, other, outC, Hout, total);
            { float* t = cur; cur = other; other = t; }
            curH = Hout;
        }
    }

    // ---- avg-pool (1x1 -> identity) + FC ----
    fc_kernel<<<(1280 + 255) / 256, 256, 0, stream>>>(
        cur, (const float*)d_in[56], (const float*)d_in[57], (float*)d_out);
}
